// RepaymentEdgePredictor_50027779064346
// MI455X (gfx1250) — matrix operand order repaired and checked by reading, not compiled, not run on hardware
//
#include <hip/hip_runtime.h>
#include <hip/hip_bf16.h>
#include <stdint.h>

// ---------------------------------------------------------------------------
// Problem constants (from the reference)
// ---------------------------------------------------------------------------
#define Nn   40000
#define Ee   800000
#define Pp   400000
#define NF   16
#define EF   8
#define Hh   128
#define EOUT 3

#define HASH_BITS 21
#define HASH_SIZE (1u << HASH_BITS)
#define HASH_MASK (HASH_SIZE - 1u)

typedef _Float16 half_t;
typedef __attribute__((ext_vector_type(16))) _Float16 v16h;
typedef __attribute__((ext_vector_type(8)))  float    v8f;

union Frag16 { v16h v; uint32_t u[8]; uint4 q[2]; };

// ---------------------------------------------------------------------------
// Generic helpers
// ---------------------------------------------------------------------------
__global__ void k_zero_f32(float* __restrict__ p, int n) {
  int i = blockIdx.x * 256 + threadIdx.x;
  if (i < n) p[i] = 0.0f;
}

__global__ void k_fill_f32(float* __restrict__ p, float v, int n) {
  int i = blockIdx.x * 256 + threadIdx.x;
  if (i < n) p[i] = v;
}

__global__ void k_f32_to_f16(const float* __restrict__ in, half_t* __restrict__ out, int n) {
  int i = blockIdx.x * 256 + threadIdx.x;
  if (i < n) out[i] = (half_t)in[i];
}

// ---------------------------------------------------------------------------
// Node projection: h0 = x @ np_w + np_b      [N,16] @ [16,128]
// ---------------------------------------------------------------------------
__global__ void k_node_proj(const float* __restrict__ x, const float* __restrict__ W,
                            const float* __restrict__ b, float* __restrict__ h0) {
  int idx = blockIdx.x * 256 + threadIdx.x;
  if (idx >= Nn * Hh) return;
  int n = idx >> 7, j = idx & 127;
  float acc = b[j];
  const float* xr = x + (size_t)n * NF;
#pragma unroll
  for (int k = 0; k < NF; ++k) acc += xr[k] * W[k * Hh + j];
  h0[idx] = acc;
}

// ---------------------------------------------------------------------------
// Degree / normalization (self loops included: deg initialized to 1)
// ---------------------------------------------------------------------------
__global__ void k_deg_accum(const int* __restrict__ dst, float* __restrict__ deg, int ne) {
  int e = blockIdx.x * 256 + threadIdx.x;
  if (e < ne) atomicAdd(&deg[dst[e]], 1.0f);
}

__global__ void k_dinv(const float* __restrict__ deg, float* __restrict__ dinv, int n) {
  int i = blockIdx.x * 256 + threadIdx.x;
  if (i < n) { float d = deg[i]; dinv[i] = d > 0.0f ? rsqrtf(d) : 0.0f; }
}

// ---------------------------------------------------------------------------
// WMMA GEMM:  out[M,NCOLS] = A[M,K](f16) @ B[K,NCOLS](f16) (+bias) (+relu)
//
// Each wave owns one full 16xNCOLS row-strip: TPR (=NCOLS/16) f32 accumulator
// tiles. Per 32-wide K-step the A fragment is loaded ONCE (two 16B global
// quads, contiguous k = 16*hfw + i order, same as B) and reused for all TPR WMMAs against B
// fragments read from LDS (B staged pre-swizzled fragment-major: each lane's
// 16 halves = 32 contiguous bytes -> two ds_load_b128). Compile-time
// K/NCOLS/flags fully unroll the pipeline and keep the epilogue branch-free.
// ---------------------------------------------------------------------------
template <int K, int NCOLS, bool RELU, bool OUT16, bool BIAS>
__global__ __launch_bounds__(256)
void k_gemm_f16(const half_t* __restrict__ A, const half_t* __restrict__ B,
                const float* __restrict__ bias, void* __restrict__ outp, int M) {
  constexpr int TPR   = NCOLS / 16;     // 16x16 tiles per row-strip
  constexpr int ELEMS = 128 * NCOLS;    // halves per 128-row B chunk
  __shared__ half_t sB[ELEMS];          // fragment-major swizzled chunk (<=32KB)

  const int tid  = threadIdx.x;
  const int wave = tid >> 5;
  const int lane = tid & 31;
  const int hfw  = lane >> 4;           // half-wave select
  const int lr   = lane & 15;

  const int strips  = M >> 4;           // one strip = 16 rows x NCOLS
  const int strip   = blockIdx.x * 8 + wave;
  const bool active = strip < strips;   // wave-uniform
  const int m0 = active ? strip << 4 : 0;

  v8f c[TPR];
#pragma unroll
  for (int nt = 0; nt < TPR; ++nt) {
    float bv = 0.0f;
    if (BIAS && active) bv = bias[nt * 16 + lr];   // bias depends on column only
#pragma unroll
    for (int v = 0; v < 8; ++v) c[nt][v] = bv;
  }

#pragma unroll
  for (int kblk = 0; kblk < K; kblk += 128) {
    __syncthreads();
    // ---- stage + swizzle one 128 x NCOLS chunk of B into LDS --------------
    // element i of lane L in k-subchunk kc, tile nt holds
    //   B[kblk + kc*32 + (L>>4)*16 + i][nt*16 + (L&15)]
    for (int t = tid; t < ELEMS; t += 256) {
      int k_local = t / NCOLS;
      int col     = t % NCOLS;
      int nt  = col >> 4, lr2 = col & 15;
      int kc  = k_local >> 5;
      int hf2 = (k_local >> 4) & 1;
      int i   = k_local & 15;
      int lane2 = hf2 * 16 + lr2;
      sB[(((kc * TPR + nt) * 32) + lane2) * 16 + i] =
          B[(size_t)(kblk + k_local) * NCOLS + col];
    }
    __syncthreads();

    if (active) {
      if (kblk + 128 < K)   // prefetch next A chunk (global_prefetch_b8)
        __builtin_prefetch(A + (size_t)(m0 + lr) * K + kblk + 128, 0, 0);
#pragma unroll
      for (int kc = 0; kc < 4; ++kc) {
        // ---- A fragment 16x32: loaded once, reused for all TPR tiles ------
        Frag16 a;
        const uint4* ap =
            (const uint4*)(A + (size_t)(m0 + lr) * K + kblk + kc * 32);
        a.q[0] = ap[2 * hfw];     // K hfw*16 .. hfw*16+7
        a.q[1] = ap[2 * hfw + 1]; // K hfw*16+8 .. hfw*16+15
#pragma unroll
        for (int nt = 0; nt < TPR; ++nt) {
          // ---- B fragment 32x16: 32 contiguous bytes per lane from LDS ----
          Frag16 b;
          const uint4* bp =
              (const uint4*)(sB + (((kc * TPR + nt) * 32) + lane) * 16);
          b.q[0] = bp[0];
          b.q[1] = bp[1];
          c[nt] = __builtin_amdgcn_wmma_f32_16x16x32_f16(
                      /*neg_a=*/false, a.v, /*neg_b=*/false, b.v,
                      /*c_mod=*/(short)0, c[nt], /*reuse_a=*/false, /*reuse_b=*/false);
        }
      }
    }
  }

  if (!active) return;
#pragma unroll
  for (int nt = 0; nt < TPR; ++nt) {
#pragma unroll
    for (int v = 0; v < 8; ++v) {
      float val = c[nt][v];
      if (RELU) val = fmaxf(val, 0.0f);
      size_t idx = (size_t)(m0 + v + hfw * 8) * NCOLS + (nt * 16 + lr);
      if (OUT16) ((half_t*)outp)[idx] = (half_t)val;
      else       ((float*)outp)[idx]  = val;
    }
  }
}

// ---------------------------------------------------------------------------
// GCN scatter: acc[d] += m[s] * dinv[s]*dinv[d]   (one wave per edge)
// ---------------------------------------------------------------------------
__global__ __launch_bounds__(256)
void k_gcn_scatter(const float* __restrict__ m, const int* __restrict__ s,
                   const int* __restrict__ d, const float* __restrict__ dinv,
                   float* __restrict__ acc, int ne) {
  int e = blockIdx.x * 8 + (threadIdx.x >> 5);
  if (e >= ne) return;
  int lane = threadIdx.x & 31;
  int si = s[e], di = d[e];
  float nrm = dinv[si] * dinv[di];
  const float* mr = m   + (size_t)si * Hh;
  float*       ar = acc + (size_t)di * Hh;
#pragma unroll
  for (int j = lane; j < Hh; j += 32) atomicAdd(&ar[j], mr[j] * nrm);
}

// Self-loop + bias (+relu):  out = acc + m*dinv^2 + b
__global__ void k_gcn_finish(const float* __restrict__ m, const float* __restrict__ dinv,
                             const float* __restrict__ bias, const float* __restrict__ acc,
                             float* __restrict__ out, int relu) {
  int idx = blockIdx.x * 256 + threadIdx.x;
  if (idx >= Nn * Hh) return;
  int n = idx >> 7, j = idx & 127;
  float di = dinv[n];
  float v = acc[idx] + m[idx] * di * di + bias[j];
  out[idx] = relu ? fmaxf(v, 0.0f) : v;
}

// ---------------------------------------------------------------------------
// Edge encoder layer 1 (K=8): t16 = relu(edge_attr @ e1_w + e1_b) as f16
// ---------------------------------------------------------------------------
__global__ void k_edge_l1(const float* __restrict__ ea, const float* __restrict__ W,
                          const float* __restrict__ b, half_t* __restrict__ t16) {
  long idx = (long)blockIdx.x * 256 + threadIdx.x;
  if (idx >= (long)Ee * Hh) return;
  long e = idx >> 7;
  int  j = (int)(idx & 127);
  float acc = b[j];
  const float* er = ea + e * EF;
#pragma unroll
  for (int k = 0; k < EF; ++k) acc += er[k] * W[k * Hh + j];
  t16[idx] = (half_t)fmaxf(acc, 0.0f);
}

// ---------------------------------------------------------------------------
// Hash join: key = src*N+dst -> max edge index (== stable-sort "last dup")
// ---------------------------------------------------------------------------
__global__ void k_hash_clear(int* __restrict__ keys, int* __restrict__ vals) {
  uint32_t i = blockIdx.x * 256 + threadIdx.x;
  if (i < HASH_SIZE) { keys[i] = -1; vals[i] = -1; }
}

__global__ void k_hash_build(const int* __restrict__ src, const int* __restrict__ dst,
                             int* __restrict__ keys, int* __restrict__ vals, int ne) {
  int e = blockIdx.x * 256 + threadIdx.x;
  if (e >= ne) return;
  uint32_t key  = (uint32_t)src[e] * (uint32_t)Nn + (uint32_t)dst[e];  // < 2^31
  uint32_t slot = (key * 2654435761u) & HASH_MASK;
  for (;;) {
    int prev = atomicCAS(&keys[slot], -1, (int)key);
    if (prev == -1 || prev == (int)key) { atomicMax(&vals[slot], e); break; }
    slot = (slot + 1) & HASH_MASK;
  }
}

// Build edge_input f16 [P,384] = [h[q_src] | h[q_dst] | epred]
__global__ __launch_bounds__(128)
void k_build_input(const float* __restrict__ h, const int* __restrict__ qs,
                   const int* __restrict__ qd, const half_t* __restrict__ emb,
                   const int* __restrict__ keys, const int* __restrict__ vals,
                   half_t* __restrict__ ein) {
  __shared__ int s_eidx;
  int p = blockIdx.x;
  int j = threadIdx.x;
  if (j == 0) {
    uint32_t key  = (uint32_t)qd[p] * (uint32_t)Nn + (uint32_t)qs[p];  // reversed pair
    uint32_t slot = (key * 2654435761u) & HASH_MASK;
    int eidx = -1;
    for (uint32_t probe = 0; probe < HASH_SIZE; ++probe) {
      int k = keys[slot];
      if (k == (int)key) { eidx = vals[slot]; break; }
      if (k == -1) break;
      slot = (slot + 1) & HASH_MASK;
    }
    s_eidx = eidx;
  }
  __syncthreads();
  int eidx = s_eidx;
  half_t* row = ein + (size_t)p * (3 * Hh);
  row[j]          = (half_t)h[(size_t)qs[p] * Hh + j];
  row[Hh + j]     = (half_t)h[(size_t)qd[p] * Hh + j];
  row[2 * Hh + j] = (eidx >= 0) ? emb[(size_t)eidx * Hh + j] : (half_t)0.0f;
}

// ---------------------------------------------------------------------------
// Output heads (K=64, tiny N): scalar dot products
// ---------------------------------------------------------------------------
__global__ void k_head1(const half_t* __restrict__ z, const float* __restrict__ w,
                        const float* __restrict__ b, float* __restrict__ out) {
  int p = blockIdx.x * 256 + threadIdx.x;
  if (p >= Pp) return;
  const half_t* zr = z + (size_t)p * 64;
  float acc = b[0];
#pragma unroll
  for (int k = 0; k < 64; ++k) acc += (float)zr[k] * w[k];
  out[p] = acc;
}

__global__ void k_head3(const half_t* __restrict__ z, const float* __restrict__ w,
                        const float* __restrict__ b, float* __restrict__ out) {
  int p = blockIdx.x * 256 + threadIdx.x;
  if (p >= Pp) return;
  const half_t* zr = z + (size_t)p * 64;
  float a0 = b[0], a1 = b[1], a2 = b[2];
#pragma unroll
  for (int k = 0; k < 64; ++k) {
    float zv = (float)zr[k];
    a0 += zv * w[k * 3 + 0];
    a1 += zv * w[k * 3 + 1];
    a2 += zv * w[k * 3 + 2];
  }
  out[(size_t)p * 3 + 0] = a0;
  out[(size_t)p * 3 + 1] = a1;
  out[(size_t)p * 3 + 2] = a2;
}

// ---------------------------------------------------------------------------
// Host orchestration
// ---------------------------------------------------------------------------
static inline int cdiv(long a, long b) { return (int)((a + b - 1) / b); }

extern "C" void kernel_launch(void* const* d_in, const int* in_sizes, int n_in,
                              void* d_out, int out_size, void* d_ws, size_t ws_size,
                              hipStream_t stream) {
  (void)in_sizes; (void)n_in; (void)out_size; (void)ws_size;
  // ---- inputs (setup_inputs dict order; params flattened in insertion order)
  const float* x      = (const float*)d_in[0];
  const int*   eidx   = (const int*)d_in[1];   // [2,E]: src=eidx, dst=eidx+E
  const float* eattr  = (const float*)d_in[2];
  const int*   qidx   = (const int*)d_in[3];   // [2,P]
  const float* np_w = (const float*)d_in[4],  *np_b = (const float*)d_in[5];
  const float* g1_w = (const float*)d_in[6],  *g1_b = (const float*)d_in[7];
  const float* g2_w = (const float*)d_in[8],  *g2_b = (const float*)d_in[9];
  const float* e1_w = (const float*)d_in[10], *e1_b = (const float*)d_in[11];
  const float* e2_w = (const float*)d_in[12], *e2_b = (const float*)d_in[13];
  const float* d1_w = (const float*)d_in[14], *d1_b = (const float*)d_in[15];
  const float* d2_w = (const float*)d_in[16], *d2_b = (const float*)d_in[17];
  const float* d3_w = (const float*)d_in[18], *d3_b = (const float*)d_in[19];
  const float* f1_w = (const float*)d_in[20], *f1_b = (const float*)d_in[21];
  const float* f2_w = (const float*)d_in[22], *f2_b = (const float*)d_in[23];
  const float* f3_w = (const float*)d_in[24], *f3_b = (const float*)d_in[25];

  const int* src = eidx;         const int* dst = eidx + Ee;
  const int* qs  = qidx;         const int* qd  = qidx + Pp;

  float* logits   = (float*)d_out;          // [P]
  float* features = (float*)d_out + Pp;     // [P,3]

  // ---- workspace carve-up (everything re-initialized every call)
  char* wp = (char*)d_ws;
  auto carve = [&](size_t bytes) -> void* {
    void* p = (void*)wp;
    wp += (bytes + 255) & ~(size_t)255;
    return p;
  };
  float*  h0     = (float*)carve((size_t)Nn * Hh * 4);   // also reused as h2
  float*  h1     = (float*)carve((size_t)Nn * Hh * 4);   // also GCN accumulator
  float*  mbuf   = (float*)carve((size_t)Nn * Hh * 4);
  half_t* hf16   = (half_t*)carve((size_t)Nn * Hh * 2);
  float*  deg    = (float*)carve((size_t)Nn * 4);
  float*  dinv   = (float*)carve((size_t)Nn * 4);
  half_t* g1w16  = (half_t*)carve((size_t)Hh * Hh * 2);
  half_t* g2w16  = (half_t*)carve((size_t)Hh * Hh * 2);
  half_t* e2w16  = (half_t*)carve((size_t)Hh * Hh * 2);
  half_t* d1w16  = (half_t*)carve((size_t)3 * Hh * Hh * 2);
  half_t* d2w16  = (half_t*)carve((size_t)Hh * 64 * 2);
  half_t* f1w16  = (half_t*)carve((size_t)3 * Hh * Hh * 2);
  half_t* f2w16  = (half_t*)carve((size_t)Hh * 64 * 2);
  half_t* t16    = (half_t*)carve((size_t)Ee * Hh * 2);  // edge-enc hidden; later reused for z1/z2
  half_t* emb16  = (half_t*)carve((size_t)Ee * Hh * 2);
  int*    hkeys  = (int*)carve((size_t)HASH_SIZE * 4);
  int*    hvals  = (int*)carve((size_t)HASH_SIZE * 4);
  half_t* ein16  = (half_t*)carve((size_t)Pp * 3 * Hh * 2);
  // z1/z2 overlay the (dead after emb16) t16 region: 153.6MB <= 204.8MB
  half_t* z1 = t16;
  half_t* z2 = t16 + (size_t)Pp * Hh;

  // ---- weight conversion to f16 (cheap; every call, ws is not persistent)
  k_f32_to_f16<<<cdiv(Hh * Hh, 256), 256, 0, stream>>>(g1_w, g1w16, Hh * Hh);
  k_f32_to_f16<<<cdiv(Hh * Hh, 256), 256, 0, stream>>>(g2_w, g2w16, Hh * Hh);
  k_f32_to_f16<<<cdiv(Hh * Hh, 256), 256, 0, stream>>>(e2_w, e2w16, Hh * Hh);
  k_f32_to_f16<<<cdiv(3 * Hh * Hh, 256), 256, 0, stream>>>(d1_w, d1w16, 3 * Hh * Hh);
  k_f32_to_f16<<<cdiv(Hh * 64, 256), 256, 0, stream>>>(d2_w, d2w16, Hh * 64);
  k_f32_to_f16<<<cdiv(3 * Hh * Hh, 256), 256, 0, stream>>>(f1_w, f1w16, 3 * Hh * Hh);
  k_f32_to_f16<<<cdiv(Hh * 64, 256), 256, 0, stream>>>(f2_w, f2w16, Hh * 64);

  // ---- node projection + degree norm
  k_node_proj<<<cdiv((long)Nn * Hh, 256), 256, 0, stream>>>(x, np_w, np_b, h0);
  k_fill_f32<<<cdiv(Nn, 256), 256, 0, stream>>>(deg, 1.0f, Nn);   // self loops
  k_deg_accum<<<cdiv(Ee, 256), 256, 0, stream>>>(dst, deg, Ee);
  k_dinv<<<cdiv(Nn, 256), 256, 0, stream>>>(deg, dinv, Nn);

  // grid: one 16-row strip per wave, 8 waves per block
  const int gcn_blocks  = cdiv(Nn / 16, 8);
  const int edge_blocks = cdiv(Ee / 16, 8);
  const int p_blocks    = cdiv(Pp / 16, 8);

  // ---- GCN layer 1:  h1 = relu(scatter(h0@g1_w) + b)
  k_f32_to_f16<<<cdiv((long)Nn * Hh, 256), 256, 0, stream>>>(h0, hf16, Nn * Hh);
  k_gemm_f16<128, 128, false, false, false><<<gcn_blocks, 256, 0, stream>>>(
      hf16, g1w16, nullptr, mbuf, Nn);
  k_zero_f32<<<cdiv((long)Nn * Hh, 256), 256, 0, stream>>>(h1, Nn * Hh);
  k_gcn_scatter<<<cdiv(Ee, 8), 256, 0, stream>>>(mbuf, src, dst, dinv, h1, Ee);
  k_gcn_finish<<<cdiv((long)Nn * Hh, 256), 256, 0, stream>>>(mbuf, dinv, g1_b, h1, h1, 1);

  // ---- GCN layer 2:  h2(=h0) = scatter(h1@g2_w) + b
  k_f32_to_f16<<<cdiv((long)Nn * Hh, 256), 256, 0, stream>>>(h1, hf16, Nn * Hh);
  k_gemm_f16<128, 128, false, false, false><<<gcn_blocks, 256, 0, stream>>>(
      hf16, g2w16, nullptr, mbuf, Nn);
  k_zero_f32<<<cdiv((long)Nn * Hh, 256), 256, 0, stream>>>(h0, Nn * Hh);
  k_gcn_scatter<<<cdiv(Ee, 8), 256, 0, stream>>>(mbuf, src, dst, dinv, h0, Ee);
  k_gcn_finish<<<cdiv((long)Nn * Hh, 256), 256, 0, stream>>>(mbuf, dinv, g2_b, h0, h0, 0);

  // ---- edge encoder: emb16 = relu(edge_attr@e1_w+b) @ e2_w + b
  k_edge_l1<<<cdiv((long)Ee * Hh, 256), 256, 0, stream>>>(eattr, e1_w, e1_b, t16);
  k_gemm_f16<128, 128, false, true, true><<<edge_blocks, 256, 0, stream>>>(
      t16, e2w16, e2_b, emb16, Ee);

  // ---- hash join (reversed-pair, last-duplicate semantics via atomicMax)
  k_hash_clear<<<cdiv(HASH_SIZE, 256), 256, 0, stream>>>(hkeys, hvals);
  k_hash_build<<<cdiv(Ee, 256), 256, 0, stream>>>(src, dst, hkeys, hvals, Ee);

  // ---- edge_input f16 [P,384]
  k_build_input<<<Pp, 128, 0, stream>>>(h0, qs, qd, emb16, hkeys, hvals, ein16);

  // ---- head d: 384->128 relu, 128->64 relu, 64->1
  k_gemm_f16<384, 128, true, true, true><<<p_blocks, 256, 0, stream>>>(
      ein16, d1w16, d1_b, z1, Pp);
  k_gemm_f16<128, 64, true, true, true><<<p_blocks, 256, 0, stream>>>(
      z1, d2w16, d2_b, z2, Pp);
  k_head1<<<cdiv(Pp, 256), 256, 0, stream>>>(z2, d3_w, d3_b, logits);

  // ---- head f: 384->128 relu, 128->64 relu, 64->3 (reuses z1/z2 after d head)
  k_gemm_f16<384, 128, true, true, true><<<p_blocks, 256, 0, stream>>>(
      ein16, f1w16, f1_b, z1, Pp);
  k_gemm_f16<128, 64, true, true, true><<<p_blocks, 256, 0, stream>>>(
      z1, f2w16, f2_b, z2, Pp);
  k_head3<<<cdiv(Pp, 256), 256, 0, stream>>>(z2, f3_w, f3_b, features);
}
